// SmoothMHA_72206990180903
// MI455X (gfx1250) — matrix operand order repaired and checked by reading, not compiled, not run on hardware
//
#include <hip/hip_runtime.h>
#include <hip/hip_bf16.h>
#include <stdint.h>

// ---------------------------------------------------------------------------
// SmoothMHA on MI455X (gfx1250): KNN via f16 WMMA distance GEMM + per-point
// neighborhood attention via f16 WMMA projections, fp32 accumulate.
// Round 3: scalarized wave id (SALU loop control / scalar-branch prefetch
// guard in the KNN hot loop), in-order-friendly prefetch issue order.
// ---------------------------------------------------------------------------

#define BB   2
#define NN   8192
#define DD   128
#define KK   12          // neighbors
#define HH   8           // heads
#define HD   16          // head dim
#define QT   16          // query tile (rows per KNN workgroup)
#define WPB  8           // waves per KNN workgroup

typedef _Float16 v16h __attribute__((ext_vector_type(16)));
typedef _Float16 v8h  __attribute__((ext_vector_type(8)));
typedef float    v8f  __attribute__((ext_vector_type(8)));

// Compiler-only reordering fence: same-wave LDS RAW is ordered by hardware
// (DS ops from a wave complete in order), we only need to stop the compiler
// from moving the tile reads above the tile writes.
#define WAVE_LDS_FENCE() asm volatile("" ::: "memory")

// A-matrix fragment (16x32 f16, one k-step): per ISA 7.12.2, lane L holds row
// M=L%16; VGPRs 0..3 = K kbase..kbase+7, VGPRs 4..7 = K kbase+16..kbase+23,
// kbase = (L<16)?0:8.  Two contiguous 16B loads from the row-major f16 row.
static __device__ inline v16h load_a_frag(const _Float16* __restrict__ row,
                                          int kstep, int kbase) {
  const v8h lo = *(const v8h*)(row + kstep * 32 + kbase);
  const v8h hi = *(const v8h*)(row + kstep * 32 + kbase + 16);
  v16h r;
#pragma unroll
  for (int i = 0; i < 8; ++i) { r[i] = lo[i]; r[i + 8] = hi[i]; }
  return r;
}

static __device__ inline v16h zero_frag() {
  v16h r;
#pragma unroll
  for (int i = 0; i < 16; ++i) r[i] = (_Float16)0.0f;
  return r;
}

// B-matrix fragment (32x16 f16): lane L holds column N=L%16; same K order as
// the A fragment: K kbase..kbase+7, kbase+16..kbase+23, kbase=khalf/2.
static __device__ inline v16h load_b_frag(const _Float16* __restrict__ colrow,
                                          int kstep, int khalf) {
  return load_a_frag(colrow, kstep, khalf >> 1);
}

static __device__ inline v8f wmma_f16(v16h a, v16h b, v8f c) {
  return __builtin_amdgcn_wmma_f32_16x16x32_f16(
      /*neg_a=*/false, a, /*neg_b=*/false, b,
      /*c_mod=*/(short)0, c, /*reuse_a=*/false, /*reuse_b=*/false);
}

// ---------------------------------------------------------------------------
// Prep: f16 copy of x + row norms
// ---------------------------------------------------------------------------
__global__ void prep_rows_kernel(const float* __restrict__ x,
                                 _Float16* __restrict__ xh,
                                 float* __restrict__ x2) {
  const int row = blockIdx.x;          // 0 .. B*N-1
  const int t   = threadIdx.x;         // 0 .. 127
  const float v = x[(size_t)row * DD + t];
  xh[(size_t)row * DD + t] = (_Float16)v;
  __shared__ float red[DD];
  red[t] = v * v;
  __syncthreads();
  for (int s = DD / 2; s > 0; s >>= 1) {
    if (t < s) red[t] += red[t + s];
    __syncthreads();
  }
  if (t == 0) x2[row] = red[0];
}

// Prep: f16 copies of the projection weights
__global__ void prep_w_kernel(const float* __restrict__ win,
                              const float* __restrict__ wout,
                              _Float16* __restrict__ whin,
                              _Float16* __restrict__ whout) {
  const int i = blockIdx.x * 256 + threadIdx.x;
  if (i < 3 * DD * DD) whin[i]  = (_Float16)win[i];
  if (i < DD * DD)     whout[i] = (_Float16)wout[i];
}

// ---------------------------------------------------------------------------
// KNN: per 16-query tile, stream all 8192 candidates as 16x16 WMMA distance
// tiles (8 waves, each owning every 8th column tile).  B fragments are
// double-buffered so next-tile loads overlap current-tile WMMA + top-k;
// wave id is scalarized so pipeline control is pure SALU.
// Per-wave LDS tile needs no workgroup barrier (same-wave LDS is in-order).
// ---------------------------------------------------------------------------
__global__ void __launch_bounds__(256)
knn_kernel(const _Float16* __restrict__ xh, const float* __restrict__ x2,
           int* __restrict__ knn_idx) {
  const int b    = blockIdx.y;
  const int q0   = blockIdx.x * QT;
  const int tid  = threadIdx.x;
  // wave index is uniform across the wave: pin it to an SGPR so the loop
  // counter and the prefetch guard compile to SALU + s_cbranch.
  const int wave = __builtin_amdgcn_readfirstlane(tid >> 5);
  const int lane = tid & 31;
  const int lm   = lane & 15;
  const int hi   = lane >> 4;            // lane half
  const int kbase = hi ? 8 : 0;
  const int khalf = hi ? 16 : 0;

  const _Float16* xb  = xh + (size_t)b * NN * DD;
  const float*    x2b = x2 + (size_t)b * NN;

  __shared__ float tileLds[WPB][QT][16];
  __shared__ float mDist[QT][WPB * KK];
  __shared__ int   mIdx [QT][WPB * KK];

  // Persistent A fragments for this tile's 16 query rows (4 k-steps = D=128)
  const _Float16* qrow = xb + (size_t)(q0 + lm) * DD;
  v16h a[4];
#pragma unroll
  for (int s = 0; s < 4; ++s) a[s] = load_a_frag(qrow, s, kbase);

  // x2 for the 8 query rows this lane's accumulator column covers
  float x2q[8];
#pragma unroll
  for (int r = 0; r < 8; ++r) x2q[r] = x2b[q0 + r + hi * 8];

  // register-resident sorted top-12 (lanes 0..15 own query rows 0..15)
  float bd[KK]; int bi[KK];
#pragma unroll
  for (int t = 0; t < KK; ++t) { bd[t] = 3.0e38f; bi[t] = 0; }

  // ---- software-pipelined candidate stream ----
  int jt = wave;
  v16h bf[4];
  float x2c;
  {
    const _Float16* crow = xb + (size_t)(jt * 16 + lm) * DD;
    x2c = x2b[jt * 16 + lm];
#pragma unroll
    for (int s = 0; s < 4; ++s) bf[s] = load_b_frag(crow, s, khalf);
  }

  while (jt < NN / 16) {
    const int c0 = jt * 16;
    const int jn = jt + WPB;

    // prefetch next tile's fragments while this tile computes
    // (x2 scalar first: VMEM completes in order, so the epilogue's wait on
    //  the previous group tolerates the full new group in flight)
    v16h bn[4]; float x2n = 0.0f;
    if (jn < NN / 16) {                  // uniform -> s_cbranch
      const _Float16* nrow = xb + (size_t)(jn * 16 + lm) * DD;
      x2n = x2b[jn * 16 + lm];
#pragma unroll
      for (int s = 0; s < 4; ++s) bn[s] = load_b_frag(nrow, s, khalf);
    }

    v8f c = {};
#pragma unroll
    for (int s = 0; s < 4; ++s) c = wmma_f16(a[s], bf[s], c);

#pragma unroll
    for (int r = 0; r < 8; ++r)
      tileLds[wave][r + hi * 8][lm] = x2q[r] + x2c - 2.0f * c[r];
    WAVE_LDS_FENCE();                    // same-wave LDS is in-order; no barrier

    if (hi == 0) {
      for (int j = 0; j < 16; ++j) {
        const float d = tileLds[wave][lm][j];
        if (d < bd[KK - 1]) {            // unrolled bubble-insert
          float cd = d; int ci = c0 + j;
#pragma unroll
          for (int t = 0; t < KK; ++t) {
            if (cd < bd[t]) {
              float td = bd[t]; int ti = bi[t];
              bd[t] = cd; bi[t] = ci; cd = td; ci = ti;
            }
          }
        }
      }
    }
    WAVE_LDS_FENCE();

#pragma unroll
    for (int s = 0; s < 4; ++s) bf[s] = bn[s];
    x2c = x2n;
    jt  = jn;
  }

  // merge 8 per-wave sorted lists -> final 12 (ascending)
  if (hi == 0) {
#pragma unroll
    for (int t = 0; t < KK; ++t) {
      mDist[lm][wave * KK + t] = bd[t];
      mIdx [lm][wave * KK + t] = bi[t];
    }
  }
  __syncthreads();
  if (tid < QT) {
    const int q = tid;
    for (int s = 0; s < KK; ++s) {
      float best = 3.0e38f; int bj = 0;
      for (int j = 0; j < WPB * KK; ++j) {
        const float d = mDist[q][j];
        if (d < best) { best = d; bj = j; }
      }
      knn_idx[((size_t)b * NN + q0 + q) * KK + s] = mIdx[q][bj];
      mDist[q][bj] = 3.0e38f;
    }
  }
}

// ---------------------------------------------------------------------------
// Attention: one wave per point.  qkv = Xg * Win^T (24 WMMA tiles, B frags
// double-buffered), VALU softmax, ctx overlays the dead q-region of the qkv
// LDS buffer, attn_out = ctx * Wout^T (8 WMMA tiles) fused with the
// sigmoid-gated neighbor reduction on accumulator registers.
// ---------------------------------------------------------------------------
__global__ void __launch_bounds__(32)
attn_kernel(const _Float16* __restrict__ xh, const int* __restrict__ knn_idx,
            const _Float16* __restrict__ whin, const float* __restrict__ in_b,
            const _Float16* __restrict__ whout, const float* __restrict__ out_b,
            const float* __restrict__ Ww, const float* __restrict__ Wb,
            float* __restrict__ out) {
  const int p    = blockIdx.x;           // 0 .. B*N-1
  const int b    = p / NN;
  const int lane = threadIdx.x;
  const int lm   = lane & 15;
  const int hi   = lane >> 4;
  const int kbase = hi ? 8 : 0;
  const int khalf = hi ? 16 : 0;

  // qkv: 16 x 384 f32.  After softmax the q columns (0..127) are dead and
  // get overlaid with ctx rows (16 x 128).
  __shared__ float qkv [16 * 3 * DD];
  __shared__ float wm  [HH * KK * KK];   // softmax weights [h][k][m]
  __shared__ float xss [16];
  __shared__ float awl [16];

  const _Float16* xb   = xh + (size_t)b * NN * DD;
  const int*      idxp = knn_idx + (size_t)p * KK;

  // Gathered-neighbor A fragments (rows 12..15 zero-padded)
  v16h a[4];
  if (lm < KK) {
    const _Float16* arow = xb + (size_t)idxp[lm] * DD;
#pragma unroll
    for (int s = 0; s < 4; ++s) a[s] = load_a_frag(arow, s, kbase);
  } else {
#pragma unroll
    for (int s = 0; s < 4; ++s) a[s] = zero_frag();
  }

  // ---- qkv projection: 24 column tiles of 16, B frags double-buffered ----
  {
    v16h bf[4];
    {
      const _Float16* brow = whin + (size_t)lm * DD;
#pragma unroll
      for (int s = 0; s < 4; ++s) bf[s] = load_b_frag(brow, s, khalf);
    }
    for (int jt = 0; jt < 24; ++jt) {
      const int j = jt * 16 + lm;
      v16h bn[4];
      if (jt + 1 < 24) {
        const _Float16* nrow = whin + (size_t)(j + 16) * DD;
#pragma unroll
        for (int s = 0; s < 4; ++s) bn[s] = load_b_frag(nrow, s, khalf);
      }
      v8f c = {};
#pragma unroll
      for (int s = 0; s < 4; ++s) c = wmma_f16(a[s], bf[s], c);
      const float bias = in_b[j];
#pragma unroll
      for (int r = 0; r < 8; ++r) qkv[(r + hi * 8) * 384 + j] = c[r] + bias;
#pragma unroll
      for (int s = 0; s < 4; ++s) bf[s] = bn[s];
    }
  }
  __syncthreads();

  // ---- scores + softmax: 96 (h,k) rows over 32 lanes ----
  for (int t = lane; t < HH * KK; t += 32) {
    const int h = t / KK, k = t % KK;
    const float* qr = &qkv[k * 384 + h * HD];
    float sc[KK]; float mx = -3.0e38f;
#pragma unroll
    for (int m = 0; m < KK; ++m) {
      const float* kr = &qkv[m * 384 + DD + h * HD];
      float s = 0.0f;
#pragma unroll
      for (int d = 0; d < HD; ++d) s += qr[d] * kr[d];
      s *= 0.25f;                        // 1/sqrt(16)
      sc[m] = s; mx = fmaxf(mx, s);
    }
    float sum = 0.0f;
#pragma unroll
    for (int m = 0; m < KK; ++m) { sc[m] = __expf(sc[m] - mx); sum += sc[m]; }
    const float inv = 1.0f / sum;
#pragma unroll
    for (int m = 0; m < KK; ++m) wm[h * (KK * KK) + k * KK + m] = sc[m] * inv;
  }
  __syncthreads();                       // q region now dead

  // ---- ctx[k][d] = sum_m w[h(d)][k][m] * v[m][d], overlaid on q region ----
  for (int u = lane; u < 16 * DD; u += 32) {
    const int k = u >> 7, d = u & (DD - 1);
    float v = 0.0f;
    if (k < KK) {
      const float* wr = &wm[(d >> 4) * (KK * KK) + k * KK];
#pragma unroll
      for (int m = 0; m < KK; ++m) v += wr[m] * qkv[m * 384 + 2 * DD + d];
    }
    qkv[k * 384 + d] = v;                // overlay: ctx row k at q slot
  }
  __syncthreads();

  // ---- head-avg weights -> sigmoid gate -> normalized neighbor weights ----
  if (lane < KK) {
    const int k = lane;
    float s = Wb[0];
#pragma unroll
    for (int m = 0; m < KK; ++m) {
      float wa = 0.0f;
#pragma unroll
      for (int h = 0; h < HH; ++h) wa += wm[h * (KK * KK) + k * KK + m];
      s += (wa * 0.125f) * Ww[m];
    }
    xss[k] = 1.0f / (1.0f + __expf(-s));
  }
  __syncthreads();
  if (lane < 16) {
    float tot = 0.0f;
#pragma unroll
    for (int k = 0; k < KK; ++k) tot += xss[k];
    awl[lane] = (lane < KK) ? xss[lane] / tot : 0.0f;
  }
  __syncthreads();

  // ---- out projection fused with gated reduction over neighbors ----
  v16h a2[4];
#pragma unroll
  for (int s = 0; s < 4; ++s) {
    const float* cr = &qkv[lm * 384 + s * 32];   // ctx overlay
#pragma unroll
    for (int i = 0; i < 8; ++i) {
      a2[s][i]     = (_Float16)cr[kbase + i];
      a2[s][i + 8] = (_Float16)cr[kbase + 16 + i];
    }
  }
  float aww[8];
#pragma unroll
  for (int r = 0; r < 8; ++r) aww[r] = awl[r + hi * 8];

  {
    v16h bf[4];
    {
      const _Float16* brow = whout + (size_t)lm * DD;
#pragma unroll
      for (int s = 0; s < 4; ++s) bf[s] = load_b_frag(brow, s, khalf);
    }
    for (int jt = 0; jt < 8; ++jt) {
      const int j = jt * 16 + lm;
      v16h bn[4];
      if (jt + 1 < 8) {
        const _Float16* nrow = whout + (size_t)(j + 16) * DD;
#pragma unroll
        for (int s = 0; s < 4; ++s) bn[s] = load_b_frag(nrow, s, khalf);
      }
      v8f c = {};
#pragma unroll
      for (int s = 0; s < 4; ++s) c = wmma_f16(a2[s], bf[s], c);
      float part = 0.0f;
#pragma unroll
      for (int r = 0; r < 8; ++r) part += aww[r] * c[r]; // rows M=r(+8), col j
      part += __shfl_xor(part, 16, 32);                  // combine lane halves
      if (hi == 0) out[(size_t)p * DD + j] = part + out_b[j];
#pragma unroll
      for (int s = 0; s < 4; ++s) bf[s] = bn[s];
    }
  }
}

// ---------------------------------------------------------------------------
extern "C" void kernel_launch(void* const* d_in, const int* in_sizes, int n_in,
                              void* d_out, int out_size, void* d_ws, size_t ws_size,
                              hipStream_t stream) {
  const float* x     = (const float*)d_in[0];   // [B,N,D]
  const float* win   = (const float*)d_in[1];   // [3D,D]
  const float* inb   = (const float*)d_in[2];   // [3D]
  const float* wout  = (const float*)d_in[3];   // [D,D]
  const float* outb  = (const float*)d_in[4];   // [D]
  const float* Ww    = (const float*)d_in[5];   // [1,K]
  const float* Wb    = (const float*)d_in[6];   // [1]
  float* out = (float*)d_out;

  // workspace carve (256B aligned)
  uint8_t* w = (uint8_t*)d_ws;
  auto carve = [&](size_t bytes) {
    uint8_t* p = w;
    w += (bytes + 255) & ~(size_t)255;
    return p;
  };
  _Float16* xh    = (_Float16*)carve((size_t)BB * NN * DD * sizeof(_Float16));
  _Float16* whin  = (_Float16*)carve((size_t)3 * DD * DD * sizeof(_Float16));
  _Float16* whout = (_Float16*)carve((size_t)DD * DD * sizeof(_Float16));
  float*    x2    = (float*)   carve((size_t)BB * NN * sizeof(float));
  int*      kidx  = (int*)     carve((size_t)BB * NN * KK * sizeof(int));
  (void)ws_size; (void)in_sizes; (void)n_in; (void)out_size;

  prep_rows_kernel<<<BB * NN, DD, 0, stream>>>(x, xh, x2);
  prep_w_kernel<<<(3 * DD * DD + 255) / 256, 256, 0, stream>>>(win, wout, whin, whout);
  knn_kernel<<<dim3(NN / QT, BB), 256, 0, stream>>>(xh, x2, kidx);
  attn_kernel<<<BB * NN, 32, 0, stream>>>(xh, kidx, whin, inb, whout, outb,
                                          Ww, Wb, out);
}
